// Decoder_74234214744419
// MI455X (gfx1250) — matrix operand order repaired and checked by reading, not compiled, not run on hardware
//
#include <hip/hip_runtime.h>
#include <math.h>
#include <stdint.h>

// ---------------------------------------------------------------------------
// MI455X (gfx1250, wave32) decoder layer.
// All GEMMs (QKV, out-proj, MoE experts) and attention run on
// v_wmma_f32_16x16x32_bf16. Contiguous global->LDS staging uses the CDNA5
// async pipe (global_load_async_to_lds_b128 + s_wait_asynccnt) to skip the
// VGPR round-trip; only transposed tiles (B of GEMM, V of attention) go
// through VGPRs. Compute-bound (~1 TFLOP matmul vs ~0.7 GB HBM traffic at
// 23.3 TB/s), so all matmul FLOPs are routed through the WMMA pipe in bf16
// with f32 accumulation.
// ---------------------------------------------------------------------------

#define Bc   2
#define Nc   2048
#define Dc   1024
#define Hc   16
#define Ec   8
#define Kc   2
#define HIDc 4096
#define DHc  64
#define Tc   (Bc * Nc)   // 4096 tokens

typedef __attribute__((ext_vector_type(16))) __bf16 bf16x16;
typedef __attribute__((ext_vector_type(8)))  __bf16 bf16x8;
typedef __attribute__((ext_vector_type(8)))  float  f32x8;

// ---------------- CDNA5 async Global->LDS copy ------------------------------
// Per-lane: LDS[ldsOff] = MEM[gaddr], 16 bytes, tracked by ASYNCcnt.
__device__ __forceinline__ void async_ld_b128(const void* lds_ptr, const void* gptr) {
  const unsigned ldsOff = (unsigned)(uintptr_t)lds_ptr;           // LDS aperture: addr[31:0]
  const unsigned long long ga = (unsigned long long)(uintptr_t)gptr;
  asm volatile("global_load_async_to_lds_b128 %0, %1, off"
               :: "v"(ldsOff), "v"(ga) : "memory");
}
__device__ __forceinline__ void wait_async0() {
  asm volatile("s_wait_asynccnt 0x0" ::: "memory");
}

// ---------------- WMMA helpers (wave32 fragment layouts, ISA 7.12.2) --------

__device__ __forceinline__ f32x8 wmma_bf16(bf16x16 a, bf16x16 b, f32x8 c) {
  return __builtin_amdgcn_wmma_f32_16x16x32_bf16(
      /*neg_a=*/false, a, /*neg_b=*/false, b,
      /*c_mod=*/(short)0, c, /*reuse_a=*/false, /*reuse_b=*/false);
}

// A fragment: 16x32 bf16, row-major source with leading dim ld.
// lane L: m = L&15, 16 contiguous k at koff (16 for upper half).
__device__ __forceinline__ bf16x16 load_frag_a(const __bf16* base, int ld, int lane) {
  const int m    = lane & 15;
  const int koff = (lane >> 4) * 16;
  const __bf16* p = base + (size_t)m * ld;
  bf16x8 lo = *(const bf16x8*)(p + koff);
  bf16x8 hi = *(const bf16x8*)(p + koff + 8);
  bf16x16 r;
#pragma unroll
  for (int i = 0; i < 8; ++i) { r[i] = lo[i]; r[i + 8] = hi[i]; }
  return r;
}

// B fragment: 32x16 bf16 stored K-contiguous per column: base is [n][k] with
// leading dim ld. lane L: n = L&15, 16 contiguous k at koff (16 for upper half).
__device__ __forceinline__ bf16x16 load_frag_b(const __bf16* base, int ld, int lane) {
  const int n    = lane & 15;
  const int koff = (lane >> 4) * 16;
  const __bf16* p = base + (size_t)n * ld + koff;
  bf16x8 lo = *(const bf16x8*)(p);
  bf16x8 hi = *(const bf16x8*)(p + 8);
  bf16x16 r;
#pragma unroll
  for (int i = 0; i < 8; ++i) { r[i] = lo[i]; r[i + 8] = hi[i]; }
  return r;
}

// ---------------- generic tiled bf16 WMMA GEMM ------------------------------
// C[M,N] = A[M,K] * B[K,N]  (+ epilogue per MODE)
// MODE 0: C = AB
// MODE 1: C = AB + bias[col]
// MODE 2: C = AB + extra[idx]                      (residual add)
// MODE 3: C += extra[row*extraStride] * (AB + bias[col])   (gated MoE accum)
template <int MODE>
__global__ __launch_bounds__(128) void gemm_bf16(
    const __bf16* __restrict__ A, const __bf16* __restrict__ Bw,
    float* __restrict__ C, int M, int Nn, int Kk,
    const float* __restrict__ bias, const float* __restrict__ extra,
    int extraStride) {
  __shared__ __align__(16) __bf16 lsA[64 * 32];   // [m][k]
  __shared__ __align__(16) __bf16 lsB[64 * 32];   // [n][k] (transposed tile)

  const int tid   = threadIdx.x;
  const int lane  = tid & 31;
  const int wave  = tid >> 5;
  const int mBase = blockIdx.y * 64;
  const int nBase = blockIdx.x * 64;
  const int wm    = (wave >> 1) * 32;
  const int wn    = (wave & 1) * 32;

  f32x8 acc[2][2] = {};

  const int am = tid & 63;         // A tile row handled by this thread
  const int ah = (tid >> 6) * 16;  // A tile k-half
  const int bk = tid & 31;         // B tile k handled by this thread
  const int bq = (tid >> 5) * 16;  // B tile n-quarter

  for (int k0 = 0; k0 < Kk; k0 += 32) {
    __syncthreads();
    {  // stage A tile via CDNA5 async Global->LDS pipe (no VGPR round-trip)
      const __bf16* g = A + (size_t)(mBase + am) * Kk + k0 + ah;
      async_ld_b128(lsA + am * 32 + ah,     g);
      async_ld_b128(lsA + am * 32 + ah + 8, g + 8);
    }
    {  // stage B tile transposed into [n][k] (transpose => VGPR path)
      const __bf16* g = Bw + (size_t)(k0 + bk) * Nn + nBase + bq;
      bf16x8 v0 = *(const bf16x8*)(g);
      bf16x8 v1 = *(const bf16x8*)(g + 8);
#pragma unroll
      for (int i = 0; i < 8; ++i) {
        lsB[(bq + i) * 32 + bk]     = v0[i];
        lsB[(bq + 8 + i) * 32 + bk] = v1[i];
      }
    }
    if (k0 + 32 < Kk) {  // speculative L2 prefetch of next A rows
      __builtin_prefetch(A + (size_t)(mBase + am) * Kk + k0 + 32, 0, 0);
    }
    wait_async0();       // own async copies landed in LDS
    __syncthreads();     // everyone's copies visible

    bf16x16 afr[2], bfr[2];
    afr[0] = load_frag_a(lsA + (wm + 0)  * 32, 32, lane);
    afr[1] = load_frag_a(lsA + (wm + 16) * 32, 32, lane);
    bfr[0] = load_frag_b(lsB + (wn + 0)  * 32, 32, lane);
    bfr[1] = load_frag_b(lsB + (wn + 16) * 32, 32, lane);
#pragma unroll
    for (int i = 0; i < 2; ++i)
#pragma unroll
      for (int j = 0; j < 2; ++j)
        acc[i][j] = wmma_bf16(afr[i], bfr[j], acc[i][j]);
  }

  // epilogue — C fragment layout: n = lane&15, rows r + 8*(lane>=16)
  const int nLane = lane & 15;
  const int rOff  = (lane >> 4) * 8;
#pragma unroll
  for (int i = 0; i < 2; ++i) {
#pragma unroll
    for (int j = 0; j < 2; ++j) {
      const int col = nBase + wn + 16 * j + nLane;
#pragma unroll
      for (int r = 0; r < 8; ++r) {
        const int row = mBase + wm + 16 * i + rOff + r;
        const size_t idx = (size_t)row * Nn + col;
        float v = acc[i][j][r];
        if (MODE == 0) {
          C[idx] = v;
        } else if (MODE == 1) {
          C[idx] = v + bias[col];
        } else if (MODE == 2) {
          C[idx] = v + extra[idx];
        } else {  // MODE 3
          const float s = extra[(size_t)row * extraStride];
          C[idx] += s * (v + bias[col]);
        }
      }
    }
  }
}

// ---------------- flash attention (1 wave / 16-row query block) -------------
__global__ __launch_bounds__(32) void attn_kernel(
    const __bf16* __restrict__ qb, const __bf16* __restrict__ kb,
    const __bf16* __restrict__ vb, __bf16* __restrict__ ao) {
  __shared__ __align__(16) __bf16 lsK[32 * 64];  // [key][d]
  __shared__ __align__(16) __bf16 lsV[64 * 32];  // [d][key] (transposed)
  __shared__ __align__(16) __bf16 lsP[16 * 32];  // probabilities as A-matrix

  const int lane = threadIdx.x;
  const int qblk = blockIdx.x;
  const int h    = blockIdx.y;
  const int b    = blockIdx.z;
  const int m0   = qblk * 16;

  const size_t headCol = (size_t)h * DHc;
  const size_t rowBase = ((size_t)b * Nc) * Dc + headCol;

  // Q fragments: chunk 0 = d[0..31], chunk 1 = d[32..63]
  bf16x16 qa[2];
  qa[0] = load_frag_a(qb + rowBase + (size_t)m0 * Dc,      Dc, lane);
  qa[1] = load_frag_a(qb + rowBase + (size_t)m0 * Dc + 32, Dc, lane);

  f32x8 o[4] = {};
  float rm[8], rs[8];
#pragma unroll
  for (int r = 0; r < 8; ++r) { rm[r] = -__builtin_inff(); rs[r] = 0.f; }

  const int nLane = lane & 15;
  const int rOff  = (lane >> 4) * 8;
  const int jend  = (m0 + 15) >> 5;  // inclusive causal block limit

  for (int j = 0; j <= jend; ++j) {
    __syncthreads();
    {  // stage K rows (lane = key) via async Global->LDS pipe
      const __bf16* g = kb + rowBase + (size_t)(j * 32 + lane) * Dc;
#pragma unroll
      for (int c8 = 0; c8 < 8; ++c8)
        async_ld_b128(lsK + lane * 64 + c8 * 8, g + c8 * 8);
    }
    {  // stage V transposed: lsV[d][key] (transpose => VGPR path)
      const __bf16* g = vb + rowBase + (size_t)(j * 32 + lane) * Dc;
#pragma unroll
      for (int c8 = 0; c8 < 8; ++c8) {
        bf16x8 v = *(const bf16x8*)(g + c8 * 8);
#pragma unroll
        for (int i = 0; i < 8; ++i) lsV[(c8 * 8 + i) * 32 + lane] = v[i];
      }
    }
    wait_async0();
    __syncthreads();

    // S = Q * K^T : two 16x16 tiles (keys 16c..16c+15), K-dim 64 = 2 wmma
    f32x8 s[2];
#pragma unroll
    for (int c = 0; c < 2; ++c) {
      bf16x16 b0 = load_frag_b(lsK + (16 * c) * 64,      64, lane);  // d 0..31
      bf16x16 b1 = load_frag_b(lsK + (16 * c) * 64 + 32, 64, lane);  // d 32..63
      f32x8 t = {};
      t = wmma_bf16(qa[0], b0, t);
      t = wmma_bf16(qa[1], b1, t);
      s[c] = t;
    }

    // scale + causal mask
#pragma unroll
    for (int c = 0; c < 2; ++c) {
      const int col = j * 32 + 16 * c + nLane;
#pragma unroll
      for (int r = 0; r < 8; ++r) {
        const int row = m0 + rOff + r;
        float v = s[c][r] * 0.125f;  // 1/sqrt(64)
        s[c][r] = (col > row) ? -__builtin_inff() : v;
      }
    }

    // online softmax update (row stats via 16-lane butterflies; wave32 halves)
#pragma unroll
    for (int r = 0; r < 8; ++r) {
      float mx = fmaxf(s[0][r], s[1][r]);
#pragma unroll
      for (int msk = 1; msk <= 8; msk <<= 1)
        mx = fmaxf(mx, __shfl_xor(mx, msk, 32));
      const float newm  = fmaxf(rm[r], mx);
      const float alpha = expf(rm[r] - newm);
      rm[r] = newm;
      float ps = 0.f;
#pragma unroll
      for (int c = 0; c < 2; ++c) {
        const float p = expf(s[c][r] - newm);
        ps += p;
        lsP[(rOff + r) * 32 + 16 * c + nLane] = (__bf16)p;
      }
#pragma unroll
      for (int msk = 1; msk <= 8; msk <<= 1)
        ps += __shfl_xor(ps, msk, 32);
      rs[r] = rs[r] * alpha + ps;
#pragma unroll
      for (int g = 0; g < 4; ++g) o[g][r] *= alpha;
    }
    __syncthreads();

    // O += P * V : P is 16x32 A-matrix, V column groups of 16
    bf16x16 pa = load_frag_a(lsP, 32, lane);
#pragma unroll
    for (int g = 0; g < 4; ++g) {
      bf16x16 vf = load_frag_b(lsV + (16 * g) * 32, 32, lane);
      o[g] = wmma_bf16(pa, vf, o[g]);
    }
  }

  // epilogue: normalize and write [b,n,h*64+d] (transpose back to [T,D])
#pragma unroll
  for (int g = 0; g < 4; ++g) {
#pragma unroll
    for (int r = 0; r < 8; ++r) {
      const int row = m0 + rOff + r;
      const int d   = 16 * g + nLane;
      const float val = o[g][r] / rs[r];
      ao[((size_t)b * Nc + row) * Dc + headCol + d] = (__bf16)val;
    }
  }
}

// ---------------- small support kernels -------------------------------------

__global__ void f32_to_bf16(const float* __restrict__ in, __bf16* __restrict__ out,
                            size_t n) {
  size_t i = (size_t)blockIdx.x * blockDim.x + threadIdx.x;
  const size_t stride = (size_t)gridDim.x * blockDim.x;
  for (; i < n; i += stride) out[i] = (__bf16)in[i];
}

__global__ void zero_f32(float* __restrict__ p, size_t n) {
  size_t i = (size_t)blockIdx.x * blockDim.x + threadIdx.x;
  const size_t stride = (size_t)gridDim.x * blockDim.x;
  for (; i < n; i += stride) p[i] = 0.f;
}

__global__ void embed_gather(const int* __restrict__ x, const float* __restrict__ emb,
                             float* __restrict__ resid) {
  const int t = blockIdx.x;
  const float* src = emb + (size_t)x[t] * Dc;
  float* dst = resid + (size_t)t * Dc;
  for (int d = threadIdx.x; d < Dc; d += blockDim.x) dst[d] = src[d];
}

__global__ __launch_bounds__(256) void layernorm_bf16(
    const float* __restrict__ in, const float* __restrict__ w,
    const float* __restrict__ bb, __bf16* __restrict__ out) {
  __shared__ float rsum[256];
  __shared__ float rsq[256];
  const int t   = blockIdx.x;
  const int tid = threadIdx.x;
  const float* px = in + (size_t)t * Dc;
  float vals[4];
  float s = 0.f, s2 = 0.f;
#pragma unroll
  for (int i = 0; i < 4; ++i) {
    const float v = px[tid + 256 * i];
    vals[i] = v; s += v; s2 += v * v;
  }
  rsum[tid] = s; rsq[tid] = s2;
  for (int off = 128; off > 0; off >>= 1) {
    __syncthreads();
    if (tid < off) { rsum[tid] += rsum[tid + off]; rsq[tid] += rsq[tid + off]; }
  }
  __syncthreads();
  const float mean = rsum[0] * (1.f / Dc);
  const float var  = rsq[0] * (1.f / Dc) - mean * mean;
  const float rstd = rsqrtf(var + 1e-5f);
  __bf16* po = out + (size_t)t * Dc;
#pragma unroll
  for (int i = 0; i < 4; ++i) {
    const int d = tid + 256 * i;
    po[d] = (__bf16)((vals[i] - mean) * rstd * w[d] + bb[d]);
  }
}

__global__ __launch_bounds__(256) void rope_kernel(
    const float* __restrict__ qf, const float* __restrict__ kf,
    __bf16* __restrict__ qo, __bf16* __restrict__ ko) {
  const int t = blockIdx.x;
  const int n = t & (Nc - 1);
  for (int p = threadIdx.x; p < Dc / 2; p += 256) {
    const int i    = p & 31;        // pair index within head (half = 32)
    const int head = p >> 5;
    const int col  = head * DHc + 2 * i;
    const float invf = powf(10000.f, -(float)i / 32.f);
    const float ang  = (float)n * invf;
    const float c = cosf(ang), sn = sinf(ang);
    const size_t base = (size_t)t * Dc + col;
    float x1 = qf[base], x2 = qf[base + 1];
    qo[base]     = (__bf16)(x1 * c - x2 * sn);
    qo[base + 1] = (__bf16)(x1 * sn + x2 * c);
    x1 = kf[base]; x2 = kf[base + 1];
    ko[base]     = (__bf16)(x1 * c - x2 * sn);
    ko[base + 1] = (__bf16)(x1 * sn + x2 * c);
  }
}

__global__ void silu_mul(const float* __restrict__ t1, const float* __restrict__ t2,
                         __bf16* __restrict__ h, size_t n) {
  size_t i = (size_t)blockIdx.x * blockDim.x + threadIdx.x;
  const size_t stride = (size_t)gridDim.x * blockDim.x;
  for (; i < n; i += stride) {
    const float a = t1[i];
    h[i] = (__bf16)((a / (1.f + expf(-a))) * t2[i]);
  }
}

// router: one wave per token; full softmax probs + top-2 gate weights
__global__ __launch_bounds__(32) void router_kernel(
    const __bf16* __restrict__ nm, const float* __restrict__ gw,
    const float* __restrict__ gb, float* __restrict__ probs,
    float* __restrict__ wgate) {
  const int t = blockIdx.x;
  const int lane = threadIdx.x;
  float acc[Ec] = {};
  for (int d = lane; d < Dc; d += 32) {
    const float xv = (float)nm[(size_t)t * Dc + d];
    const float* g = gw + (size_t)d * Ec;
#pragma unroll
    for (int e = 0; e < Ec; ++e) acc[e] += xv * g[e];
  }
#pragma unroll
  for (int e = 0; e < Ec; ++e)
#pragma unroll
    for (int msk = 1; msk <= 16; msk <<= 1)
      acc[e] += __shfl_xor(acc[e], msk, 32);
  if (lane == 0) {
    float l[Ec], mx = -__builtin_inff();
#pragma unroll
    for (int e = 0; e < Ec; ++e) { l[e] = acc[e] + gb[e]; mx = fmaxf(mx, l[e]); }
    float se = 0.f, pr[Ec];
#pragma unroll
    for (int e = 0; e < Ec; ++e) { pr[e] = expf(l[e] - mx); se += pr[e]; }
#pragma unroll
    for (int e = 0; e < Ec; ++e) probs[(size_t)t * Ec + e] = pr[e] / se;
    int i0 = 0;
#pragma unroll
    for (int e = 1; e < Ec; ++e) if (l[e] > l[i0]) i0 = e;
    int i1 = (i0 == 0) ? 1 : 0;
#pragma unroll
    for (int e = 0; e < Ec; ++e)
      if (e != i0 && l[e] > l[i1]) i1 = e;
    const float p1 = expf(l[i1] - l[i0]);
    const float inv = 1.f / (1.f + p1);
#pragma unroll
    for (int e = 0; e < Ec; ++e) wgate[(size_t)t * Ec + e] = 0.f;
    wgate[(size_t)t * Ec + i0] = inv;
    wgate[(size_t)t * Ec + i1] = p1 * inv;
  }
}

// deterministic fixed-order aux-loss reduction (no float atomics)
__global__ __launch_bounds__(256) void aux_kernel(
    const float* __restrict__ probs, const float* __restrict__ wgate,
    float* __restrict__ outAux) {
  __shared__ float simp[256 * Ec];
  __shared__ float sld[256 * Ec];
  const int tid = threadIdx.x;
  float imp[Ec] = {}, ld[Ec] = {};
  for (int t = tid; t < Tc; t += 256) {
#pragma unroll
    for (int e = 0; e < Ec; ++e) {
      imp[e] += probs[(size_t)t * Ec + e];
      ld[e]  += (wgate[(size_t)t * Ec + e] > 0.f) ? 1.f : 0.f;
    }
  }
#pragma unroll
  for (int e = 0; e < Ec; ++e) { simp[tid * Ec + e] = imp[e]; sld[tid * Ec + e] = ld[e]; }
  for (int off = 128; off > 0; off >>= 1) {
    __syncthreads();
    if (tid < off)
#pragma unroll
      for (int e = 0; e < Ec; ++e) {
        simp[tid * Ec + e] += simp[(tid + off) * Ec + e];
        sld[tid * Ec + e]  += sld[(tid + off) * Ec + e];
      }
  }
  __syncthreads();
  if (tid == 0) {
    float aux = 0.f;
#pragma unroll
    for (int e = 0; e < Ec; ++e)
      aux += (simp[e] / (float)Tc) * (sld[e] / (float)(Tc * Kc));
    outAux[0] = aux * (float)Ec;
  }
}

// ---------------------------------------------------------------------------

extern "C" void kernel_launch(void* const* d_in, const int* in_sizes, int n_in,
                              void* d_out, int out_size, void* d_ws, size_t ws_size,
                              hipStream_t stream) {
  (void)in_sizes; (void)n_in; (void)out_size; (void)ws_size;
  const int*   x        = (const int*)  d_in[0];
  const float* emb      = (const float*)d_in[1];
  const float* ln1_w    = (const float*)d_in[2];
  const float* ln1_b    = (const float*)d_in[3];
  const float* ln2_w    = (const float*)d_in[4];
  const float* ln2_b    = (const float*)d_in[5];
  const float* q_proj   = (const float*)d_in[6];
  const float* k_proj   = (const float*)d_in[7];
  const float* v_proj   = (const float*)d_in[8];
  const float* out_proj = (const float*)d_in[9];
  const float* gate_w   = (const float*)d_in[10];
  const float* gate_b   = (const float*)d_in[11];
  const float* w1       = (const float*)d_in[12];
  const float* b1       = (const float*)d_in[13];
  const float* w2       = (const float*)d_in[14];
  const float* b2       = (const float*)d_in[15];
  const float* w3       = (const float*)d_in[16];
  const float* b3       = (const float*)d_in[17];
  float* out = (float*)d_out;

  // workspace arena (256B aligned slices)
  char* wsp = (char*)d_ws;
  size_t off = 0;
  auto alloc = [&](size_t bytes) -> char* {
    char* p = wsp + off;
    off += (bytes + 255) & ~(size_t)255;
    return p;
  };
  const size_t DD  = (size_t)Dc * Dc;
  const size_t EDH = (size_t)Ec * Dc * HIDc;
  __bf16* wqB = (__bf16*)alloc(DD * 2);
  __bf16* wkB = (__bf16*)alloc(DD * 2);
  __bf16* wvB = (__bf16*)alloc(DD * 2);
  __bf16* woB = (__bf16*)alloc(DD * 2);
  __bf16* w1B = (__bf16*)alloc(EDH * 2);
  __bf16* w2B = (__bf16*)alloc(EDH * 2);
  __bf16* w3B = (__bf16*)alloc(EDH * 2);
  float*  resid = (float*) alloc((size_t)Tc * Dc * 4);
  __bf16* pn    = (__bf16*)alloc((size_t)Tc * Dc * 2);
  float*  qf    = (float*) alloc((size_t)Tc * Dc * 4);
  float*  kf    = (float*) alloc((size_t)Tc * Dc * 4);
  float*  vf    = (float*) alloc((size_t)Tc * Dc * 4);
  __bf16* qB    = (__bf16*)alloc((size_t)Tc * Dc * 2);
  __bf16* kB    = (__bf16*)alloc((size_t)Tc * Dc * 2);
  __bf16* vB    = (__bf16*)alloc((size_t)Tc * Dc * 2);
  __bf16* aoB   = (__bf16*)alloc((size_t)Tc * Dc * 2);
  float*  atto  = (float*) alloc((size_t)Tc * Dc * 4);
  __bf16* nmB   = (__bf16*)alloc((size_t)Tc * Dc * 2);
  float*  probs = (float*) alloc((size_t)Tc * Ec * 4);
  float*  wgate = (float*) alloc((size_t)Tc * Ec * 4);
  float*  t1    = (float*) alloc((size_t)Tc * HIDc * 4);
  float*  t2    = (float*) alloc((size_t)Tc * HIDc * 4);
  __bf16* hB    = (__bf16*)alloc((size_t)Tc * HIDc * 2);

  // 1) weight conversion f32 -> bf16
  f32_to_bf16<<<2048, 256, 0, stream>>>(q_proj,   wqB, DD);
  f32_to_bf16<<<2048, 256, 0, stream>>>(k_proj,   wkB, DD);
  f32_to_bf16<<<2048, 256, 0, stream>>>(v_proj,   wvB, DD);
  f32_to_bf16<<<2048, 256, 0, stream>>>(out_proj, woB, DD);
  f32_to_bf16<<<16384, 256, 0, stream>>>(w1, w1B, EDH);
  f32_to_bf16<<<16384, 256, 0, stream>>>(w2, w2B, EDH);
  f32_to_bf16<<<16384, 256, 0, stream>>>(w3, w3B, EDH);

  // 2) embedding + LN1
  embed_gather<<<Tc, 256, 0, stream>>>(x, emb, resid);
  layernorm_bf16<<<Tc, 256, 0, stream>>>(resid, ln1_w, ln1_b, pn);

  // 3) QKV projections (WMMA)
  dim3 gProj(Dc / 64, Tc / 64);
  gemm_bf16<0><<<gProj, 128, 0, stream>>>(pn, wqB, qf, Tc, Dc, Dc, nullptr, nullptr, 0);
  gemm_bf16<0><<<gProj, 128, 0, stream>>>(pn, wkB, kf, Tc, Dc, Dc, nullptr, nullptr, 0);
  gemm_bf16<0><<<gProj, 128, 0, stream>>>(pn, wvB, vf, Tc, Dc, Dc, nullptr, nullptr, 0);

  // 4) RoPE (q,k) + bf16 cast of v
  rope_kernel<<<Tc, 256, 0, stream>>>(qf, kf, qB, kB);
  f32_to_bf16<<<2048, 256, 0, stream>>>(vf, vB, (size_t)Tc * Dc);

  // 5) causal flash attention (WMMA), writes [T,D] bf16
  dim3 gAtt(Nc / 16, Hc, Bc);
  attn_kernel<<<gAtt, 32, 0, stream>>>(qB, kB, vB, aoB);

  // 6) out-proj + residual, LN2
  gemm_bf16<2><<<gProj, 128, 0, stream>>>(aoB, woB, atto, Tc, Dc, Dc, nullptr, resid, 0);
  layernorm_bf16<<<Tc, 256, 0, stream>>>(atto, ln2_w, ln2_b, nmB);

  // 7) router (top-2 softmax gates + full softmax probs for aux)
  router_kernel<<<Tc, 32, 0, stream>>>(nmB, gate_w, gate_b, probs, wgate);

  // 8) MoE experts: dense per-expert GEMMs, gate-weighted accumulate into out.
  //    (unselected experts contribute with weight 0 — same math as reference)
  zero_f32<<<2048, 256, 0, stream>>>(out, (size_t)Tc * Dc);
  dim3 gUp(HIDc / 64, Tc / 64);
  dim3 gDn(Dc / 64, Tc / 64);
  for (int e = 0; e < Ec; ++e) {
    const __bf16* w1e = w1B + (size_t)e * Dc * HIDc;
    const __bf16* w2e = w2B + (size_t)e * Dc * HIDc;
    const __bf16* w3e = w3B + (size_t)e * HIDc * Dc;
    gemm_bf16<1><<<gUp, 128, 0, stream>>>(nmB, w1e, t1, Tc, HIDc, Dc,
                                          b1 + (size_t)e * HIDc, nullptr, 0);
    gemm_bf16<1><<<gUp, 128, 0, stream>>>(nmB, w2e, t2, Tc, HIDc, Dc,
                                          b2 + (size_t)e * HIDc, nullptr, 0);
    silu_mul<<<8192, 256, 0, stream>>>(t1, t2, hB, (size_t)Tc * HIDc);
    gemm_bf16<3><<<gDn, 128, 0, stream>>>(hB, w3e, out, Tc, Dc, HIDc,
                                          b3 + (size_t)e * Dc, wgate + e, Ec);
  }

  // 9) load-balance aux loss -> out[T*D]
  aux_kernel<<<1, 256, 0, stream>>>(probs, wgate, out + (size_t)Tc * Dc);
}
